// HierarchicalVectorQuantizer_48567490183829
// MI455X (gfx1250) — compile-verified, matrix operand order checked by reading, not run on hardware
//
#include <hip/hip_runtime.h>
#include <hip/hip_bf16.h>

// ---------------------------------------------------------------------------
// HierarchicalVectorQuantizer for MI455X (gfx1250, wave32, WMMA, async-LDS)
//   latent [8,512,4096] f32, proj_w [32,512], proj_b [32], codebooks [32,1024,32]
//   out: quantized [8,32,4096] f32  ++  indices [32,8,4096] (as f32)
// Pipeline:
//   prep : codebooks f32 -> bf16 (ws) + per-code ||c||^2 (ws)      [once]
//   proj : bf16 WMMA GEMM  x = latent^T * W^T + b  -> ws (f32)
//   rvq  : 32 rounds; codebook staged via global_load_async_to_lds_b128 into
//          double-buffered LDS (overlapped with compute, s_wait_asynccnt);
//          distances = ||c||^2 - 2 r.c via v_wmma_f32_16x16x32_bf16 with the
//          accumulator seeded with ||c||^2; argmin via shfl_xor reduce.
// ---------------------------------------------------------------------------

typedef __attribute__((ext_vector_type(16))) __bf16 v16bf;
typedef __attribute__((ext_vector_type(8)))  float  v8f;

struct U16x16 { uint4 lo; uint4 hi; };   // 32 bytes, matches v16bf

static __device__ __forceinline__ unsigned short f32_to_bf16_rne(float f) {
    unsigned int u = __float_as_uint(f);
    u += 0x7FFFu + ((u >> 16) & 1u);          // round-to-nearest-even
    return (unsigned short)(u >> 16);
}
static __device__ __forceinline__ float bfbits_lo(unsigned int u) {
    return __uint_as_float((u & 0xFFFFu) << 16);
}
static __device__ __forceinline__ float bfbits_hi(unsigned int u) {
    return __uint_as_float(u & 0xFFFF0000u);
}
static __device__ __forceinline__ ushort4 f4_to_bf4(float4 v) {
    ushort4 r;
    r.x = f32_to_bf16_rne(v.x); r.y = f32_to_bf16_rne(v.y);
    r.z = f32_to_bf16_rne(v.z); r.w = f32_to_bf16_rne(v.w);
    return r;
}

// CDNA5 async global->LDS copy, 16B per lane (tracked by ASYNCcnt).
static __device__ __forceinline__ void async_copy_b128(unsigned lds_off, const void* gptr) {
    asm volatile("global_load_async_to_lds_b128 %0, %1, off"
                 :: "v"(lds_off), "v"((unsigned long long)gptr) : "memory");
}
static __device__ __forceinline__ void wait_async_le17() {
    asm volatile("s_wait_asynccnt 0x11" ::: "memory");
}
static __device__ __forceinline__ void wait_async_0() {
    asm volatile("s_wait_asynccnt 0x0" ::: "memory");
}

#define BT      32768      // B*T
#define CDIM    512
#define DDIM    32
#define KCODES  1024
#define NCB     32
#define LATP    520        // padded LDS row stride (ushorts) for 512-c rows
#define CBP     40         // padded LDS stride per code (ushorts): 80B, 16B-mult

// ====================== Kernel 0: codebook prep (once) ======================
// cbbf[cb][code][dim] = bf16(codebooks); cnorms[cb][code] = sum(bf16(c)^2)
__launch_bounds__(256, 1)
__global__ void hvq_prep_kernel(const float* __restrict__ codebooks,
                                unsigned short* __restrict__ cbbf,
                                float* __restrict__ cnorms) {
    int code = blockIdx.x * 256 + threadIdx.x;       // 0..32767 (cb*1024+k)
    const float* src = codebooks + (size_t)code * 32;
    unsigned short* dst = cbbf + (size_t)code * 32;
    float s = 0.f;
#pragma unroll
    for (int p = 0; p < 8; ++p) {
        float4 v = *(const float4*)(src + p * 4);
        ushort4 h = f4_to_bf4(v);
        float a0 = __uint_as_float(((unsigned)h.x) << 16);
        float a1 = __uint_as_float(((unsigned)h.y) << 16);
        float a2 = __uint_as_float(((unsigned)h.z) << 16);
        float a3 = __uint_as_float(((unsigned)h.w) << 16);
        s += a0 * a0 + a1 * a1 + a2 * a2 + a3 * a3;
        *(ushort4*)(dst + p * 4) = h;
    }
    cnorms[code] = s;
}

// ============================ Kernel 1: projection ==========================
// x[token][d] = sum_c latent[b][c][t] * w[d][c] + bias[d]
// 128 threads = 4 waves; one 16-token tile per wave; 512 blocks.
__launch_bounds__(128, 1)
__global__ void hvq_proj_kernel(const float* __restrict__ latent,
                                const float* __restrict__ w,
                                const float* __restrict__ bias,
                                float* __restrict__ x) {
    __shared__ __align__(16) unsigned short latA[4 * 16 * LATP]; // 4 tiles [16 t][520 c]
    __shared__ __align__(16) unsigned short wB[DDIM * LATP];     // [32 d][520 c]

    const int  tid     = threadIdx.x;
    const long tokBase = (long)blockIdx.x * 64;   // 64 tokens per WG, same b
    const int  b       = (int)(tokBase >> 12);
    const int  tIn     = (int)(tokBase & 4095);
    const float* latBase = latent + ((long)b * CDIM) * 4096 + tIn;

    for (int q = tid; q < CDIM * 16; q += 128) {
        int c  = q >> 4;
        int j4 = (q & 15) << 2;                  // 4 consecutive tokens
        float4 v = *(const float4*)(latBase + (long)c * 4096 + j4);
        int tile = j4 >> 4, tl = j4 & 15;
        unsigned short* p = &latA[tile * (16 * LATP) + tl * LATP + c];
        p[0]        = f32_to_bf16_rne(v.x);
        p[LATP]     = f32_to_bf16_rne(v.y);
        p[2 * LATP] = f32_to_bf16_rne(v.z);
        p[3 * LATP] = f32_to_bf16_rne(v.w);
    }
    for (int q = tid; q < (DDIM * CDIM) / 4; q += 128) {
        int d = q >> 7;
        int c = (q & 127) << 2;
        float4 v = *(const float4*)(w + (long)q * 4);
        *(ushort4*)&wB[d * LATP + c] = f4_to_bf4(v);
    }
    __syncthreads();

    const int wave = tid >> 5, lane = tid & 31;
    const int m = lane & 15, hi = lane >> 4;
    const unsigned short* aRow = &latA[wave * (16 * LATP) + m * LATP];

    v8f acc0 = {}, acc1 = {};
    for (int kc = 0; kc < 16; ++kc) {
        int ko = kc * 32 + hi * 8;
        U16x16 ap;
        ap.lo = *(const uint4*)(aRow + ko);
        ap.hi = *(const uint4*)(aRow + ko + 16);
        v16bf a = __builtin_bit_cast(v16bf, ap);

        U16x16 bp0, bp1;
        const unsigned short* b0 = &wB[m * LATP + ko];
        const unsigned short* b1 = &wB[(16 + m) * LATP + ko];
        bp0.lo = *(const uint4*)(b0); bp0.hi = *(const uint4*)(b0 + 16);
        bp1.lo = *(const uint4*)(b1); bp1.hi = *(const uint4*)(b1 + 16);
        v16bf vb0 = __builtin_bit_cast(v16bf, bp0);
        v16bf vb1 = __builtin_bit_cast(v16bf, bp1);

        acc0 = __builtin_amdgcn_wmma_f32_16x16x32_bf16(false, a, false, vb0,
                                                       (short)0, acc0, false, false);
        acc1 = __builtin_amdgcn_wmma_f32_16x16x32_bf16(false, a, false, vb1,
                                                       (short)0, acc1, false, false);
    }

    const long tileTok = tokBase + wave * 16;
    float bia0 = bias[m], bia1 = bias[16 + m];
#pragma unroll
    for (int j = 0; j < 8; ++j) {
        long tok = tileTok + j + 8 * hi;
        x[tok * 32 + m]      = acc0[j] + bia0;
        x[tok * 32 + 16 + m] = acc1[j] + bia1;
    }
}

// ============================ Kernel 2: RVQ loop ============================
// 256 threads = 8 waves = 128 tokens per WG; 256 blocks.
// Double-buffered LDS codebook, filled by async-to-LDS DMA overlapped with WMMA.
__launch_bounds__(256, 1)
__global__ void hvq_rvq_kernel(const unsigned short* __restrict__ cbbf, // [32][1024][32] bf16
                               const float* __restrict__ cnorms,        // [32][1024]
                               const float* __restrict__ x,
                               float* __restrict__ qout,     // [B][32][T]
                               float* __restrict__ idxout) { // [32][B*T]
    __shared__ __align__(16) unsigned short cbl[2][KCODES * CBP]; // 2 x 80 KB
    __shared__ __align__(16) float cnl[2][KCODES];                // 2 x 4 KB
    __shared__ int idxbuf[8 * 16];

    const int tid = threadIdx.x;
    const int wave = tid >> 5, lane = tid & 31;
    const int m = lane & 15, hi = lane >> 4;
    const int tokBase = blockIdx.x * 128;
    const int token   = tokBase + wave * 16 + m;   // this lane's row token

    const unsigned cblL[2] = { (unsigned)(size_t)&cbl[0][0], (unsigned)(size_t)&cbl[1][0] };
    const unsigned cnlL[2] = { (unsigned)(size_t)&cnl[0][0], (unsigned)(size_t)&cnl[1][0] };

    // Issue one round's codebook + cnorm into LDS buffer `buf` (17 async ops/wave).
    auto issue_group = [&](int cb, int buf) {
        const unsigned short* gsrc = cbbf + (size_t)cb * (KCODES * DDIM);
        unsigned lcb = cblL[buf];
#pragma unroll
        for (int r = 0; r < 16; ++r) {
            int g = r * 256 + tid;                // 16B granule id, 0..4095
            int code = g >> 2, part = g & 3;      // 4 granules per 64B code
            async_copy_b128(lcb + (unsigned)(code * (CBP * 2) + part * 16),
                            gsrc + (size_t)g * 8);
        }
        async_copy_b128(cnlL[buf] + (unsigned)(tid * 16),
                        cnorms + (size_t)cb * KCODES + tid * 4);
    };

    // Residual in native A-fragment layout: e<8 -> dim hi*8+e ; e>=8 -> 16+hi*8+(e-8)
    float res[16], qsum[16];
    {
        const float* xb = x + (long)token * 32;
        float4 r0 = *(const float4*)(xb + hi * 8);
        float4 r1 = *(const float4*)(xb + hi * 8 + 4);
        float4 r2 = *(const float4*)(xb + 16 + hi * 8);
        float4 r3 = *(const float4*)(xb + 20 + hi * 8);
        res[0]=r0.x; res[1]=r0.y; res[2]=r0.z;  res[3]=r0.w;
        res[4]=r1.x; res[5]=r1.y; res[6]=r1.z;  res[7]=r1.w;
        res[8]=r2.x; res[9]=r2.y; res[10]=r2.z; res[11]=r2.w;
        res[12]=r3.x; res[13]=r3.y; res[14]=r3.z; res[15]=r3.w;
#pragma unroll
        for (int e = 0; e < 16; ++e) qsum[e] = 0.f;
    }

    issue_group(0, 0);

    for (int cb = 0; cb < NCB; ++cb) {
        const int buf = cb & 1;
        if (cb + 1 < NCB) {
            issue_group(cb + 1, buf ^ 1);  // overlap next round's DMA with compute
            wait_async_le17();             // in-order: <=17 left => round cb landed
        } else {
            wait_async_0();
        }
        __syncthreads();

        // Pack A = -2 * residual (so WMMA gives cnorm - 2*dot directly).
        U16x16 apk;
        {
            unsigned int au[8];
#pragma unroll
            for (int p = 0; p < 8; ++p) {
                unsigned int l0 = f32_to_bf16_rne(-2.0f * res[2 * p]);
                unsigned int l1 = f32_to_bf16_rne(-2.0f * res[2 * p + 1]);
                au[p] = l0 | (l1 << 16);
            }
            apk.lo = make_uint4(au[0], au[1], au[2], au[3]);
            apk.hi = make_uint4(au[4], au[5], au[6], au[7]);
        }
        v16bf a = __builtin_bit_cast(v16bf, apk);

        float minv[8]; int mini[8];
#pragma unroll
        for (int j = 0; j < 8; ++j) { minv[j] = 3.4e38f; mini[j] = 0; }

#pragma unroll 4
        for (int chunk = 0; chunk < 64; ++chunk) {
            int code = chunk * 16 + m;
            const unsigned short* cRow = &cbl[buf][code * CBP + hi * 8];
            U16x16 bp;
            bp.lo = *(const uint4*)(cRow);
            bp.hi = *(const uint4*)(cRow + 16);
            v16bf vb = __builtin_bit_cast(v16bf, bp);
            float cn = cnl[buf][code];
            v8f acc = {cn, cn, cn, cn, cn, cn, cn, cn};
            acc = __builtin_amdgcn_wmma_f32_16x16x32_bf16(false, a, false, vb,
                                                          (short)0, acc, false, false);
#pragma unroll
            for (int j = 0; j < 8; ++j) {
                if (acc[j] < minv[j]) { minv[j] = acc[j]; mini[j] = code; }
            }
        }

        // Min-reduce across the 16 lanes of each half (C layout: row j+8*hi, col m).
#pragma unroll
        for (int j = 0; j < 8; ++j) {
            float v = minv[j]; int ix = mini[j];
#pragma unroll
            for (int msk = 1; msk < 16; msk <<= 1) {
                float ov = __shfl_xor(v, msk, 32);
                int   oi = __shfl_xor(ix, msk, 32);
                if (ov < v || (ov == v && oi < ix)) { v = ov; ix = oi; }
            }
            minv[j] = v; mini[j] = ix;
        }
        if (m == 0) {       // lane 0 -> rows 0..7, lane 16 -> rows 8..15
            int base = wave * 16 + hi * 8;
#pragma unroll
            for (int j = 0; j < 8; ++j) idxbuf[base + j] = mini[j];
        }

        // Gather winning code (bf16 from LDS) and update residual / qsum.
        int myidx = idxbuf[wave * 16 + m];
        const unsigned short* qRow = &cbl[buf][myidx * CBP + hi * 8];
        uint4 qlo = *(const uint4*)(qRow);
        uint4 qhi = *(const uint4*)(qRow + 16);
        unsigned int uu[8] = {qlo.x, qlo.y, qlo.z, qlo.w, qhi.x, qhi.y, qhi.z, qhi.w};
#pragma unroll
        for (int p = 0; p < 8; ++p) {
            float q0 = bfbits_lo(uu[p]), q1 = bfbits_hi(uu[p]);
            res[2 * p]     -= q0;  qsum[2 * p]     += q0;
            res[2 * p + 1] -= q1;  qsum[2 * p + 1] += q1;
        }

        if (!hi)  // indices[cb][b][t] == [cb][flat token]
            idxout[(long)cb * BT + token] = (float)myidx;

        __syncthreads();   // all waves done reading buf before it is refilled
    }

    // quantized forward value == qsum; write transposed [B][32][T]
    const int bb = token >> 12, tt = token & 4095;
#pragma unroll
    for (int e = 0; e < 16; ++e) {
        int d = (e < 8) ? (hi * 8 + e) : (16 + hi * 8 + (e - 8));
        qout[(long)bb * (32 * 4096) + (long)d * 4096 + tt] = qsum[e];
    }
}

// ================================ Launch ===================================
extern "C" void kernel_launch(void* const* d_in, const int* in_sizes, int n_in,
                              void* d_out, int out_size, void* d_ws, size_t ws_size,
                              hipStream_t stream) {
    (void)in_sizes; (void)n_in; (void)out_size; (void)ws_size;
    const float* latent    = (const float*)d_in[0];
    const float* proj_w    = (const float*)d_in[1];
    const float* proj_b    = (const float*)d_in[2];
    const float* codebooks = (const float*)d_in[3];

    float* qout   = (float*)d_out;                 // 8*32*4096 floats
    float* idxout = (float*)d_out + (8 * 32 * 4096);

    // Workspace layout: x (4 MB) | codebooks bf16 (2 MB) | cnorms (128 KB)
    float*          xws  = (float*)d_ws;
    unsigned short* cbbf = (unsigned short*)((char*)d_ws + (4u << 20));
    float*          cnw  = (float*)((char*)d_ws + (6u << 20));

    hvq_prep_kernel<<<128, 256, 0, stream>>>(codebooks, cbbf, cnw);
    hvq_proj_kernel<<<512, 128, 0, stream>>>(latent, proj_w, proj_b, xws);
    hvq_rvq_kernel<<<256, 256, 0, stream>>>(cbbf, cnw, xws, qout, idxout);
}
